// TransformerDecoderLayer_33921651704168
// MI455X (gfx1250) — matrix operand order repaired and checked by reading, not compiled, not run on hardware
//
#include <hip/hip_runtime.h>
#include <hip/hip_bf16.h>

// Problem constants (match reference)
#define B_  4
#define S_  1024
#define D_  1024
#define H_  16
#define DH_ 64
#define F_  4096
#define R_  (B_*S_)   // 4096 token rows

typedef __attribute__((ext_vector_type(16))) __bf16 v16bf;
typedef __attribute__((ext_vector_type(8)))  float  v8f;
typedef int v4i_b __attribute__((vector_size(16)));   // matches builtin's V4i

// ---------- CDNA5 async global->LDS copy (guarded; sync fallback) ----------
#if defined(__gfx1250__) && \
    __has_builtin(__builtin_amdgcn_global_load_async_to_lds_b128) && \
    __has_builtin(__builtin_amdgcn_s_wait_asynccnt)
#define USE_ASYNC_LDS 1
#else
#define USE_ASYNC_LDS 0
#endif

static __device__ __forceinline__ void async_cp16(void* lds, const void* gsrc) {
#if USE_ASYNC_LDS
    __builtin_amdgcn_global_load_async_to_lds_b128(
        (v4i_b __attribute__((address_space(1)))*)(unsigned long long)gsrc,
        (v4i_b __attribute__((address_space(3)))*)(unsigned int)(unsigned long long)lds,
        0, 0);
#else
    *(uint4*)lds = *(const uint4*)gsrc;
#endif
}
static __device__ __forceinline__ void wait_async_lds() {
#if USE_ASYNC_LDS
    __builtin_amdgcn_s_wait_asynccnt(0);
#endif
}

// ---------- scalar helpers ----------
static __device__ __forceinline__ unsigned short f2bf(float f) {
    union { float f; unsigned u; } c; c.f = f;
    unsigned u = c.u;
    u += 0x7FFFu + ((u >> 16) & 1u);     // round-to-nearest-even
    return (unsigned short)(u >> 16);
}
static __device__ __forceinline__ __bf16 us2bf(unsigned short s) {
    union { unsigned short s; __bf16 h; } c; c.s = s; return c.h;
}

// ---------- fp32 -> bf16 elementwise convert ----------
__global__ __launch_bounds__(256) void convert_f32_bf16(
    const float* __restrict__ in, unsigned short* __restrict__ out, int n)
{
    int i = (blockIdx.x * 256 + threadIdx.x) * 4;
    if (i + 3 < n) {
        float4 v = *(const float4*)(in + i);
        out[i + 0] = f2bf(v.x);
        out[i + 1] = f2bf(v.y);
        out[i + 2] = f2bf(v.z);
        out[i + 3] = f2bf(v.w);
    }
}

// ---------- LayerNorm: fp32 row -> bf16 row ----------
__global__ __launch_bounds__(256) void layernorm_bf16(
    const float* __restrict__ x, const float* __restrict__ g,
    const float* __restrict__ b, unsigned short* __restrict__ out)
{
    __shared__ float s1[256], s2[256];
    const int row = blockIdx.x;
    const int t = threadIdx.x;
    const float* xr = x + (size_t)row * D_;
    float4 v = *(const float4*)(xr + t * 4);
    s1[t] = v.x + v.y + v.z + v.w;
    s2[t] = v.x*v.x + v.y*v.y + v.z*v.z + v.w*v.w;
    __syncthreads();
    for (int o = 128; o > 0; o >>= 1) {
        if (t < o) { s1[t] += s1[t + o]; s2[t] += s2[t + o]; }
        __syncthreads();
    }
    float mu  = s1[0] * (1.0f / D_);
    float var = s2[0] * (1.0f / D_) - mu * mu;
    float rs  = rsqrtf(var + 1e-5f);
    float4 gv = *(const float4*)(g + t * 4);
    float4 bv = *(const float4*)(b + t * 4);
    unsigned short* orow = out + (size_t)row * D_ + t * 4;
    orow[0] = f2bf((v.x - mu) * rs * gv.x + bv.x);
    orow[1] = f2bf((v.y - mu) * rs * gv.y + bv.y);
    orow[2] = f2bf((v.z - mu) * rs * gv.z + bv.z);
    orow[3] = f2bf((v.w - mu) * rs * gv.w + bv.w);
}

// ---------- bf16 WMMA GEMM: out = act(A @ W + bias) [+ resid] ----------
// block tile 128x128, BK=64, double-buffered LDS, async global->LDS prefetch.
// 256 threads = 8 waves (2 in M x 4 in N); wave: 64x32 = 4x2 tiles of 16x16.
template<int RELU, int OUTBF, int RES>
__global__ __launch_bounds__(256) void gemm_bf16_wmma(
    const unsigned short* __restrict__ A, const unsigned short* __restrict__ W,
    const float* __restrict__ bias, const float* __restrict__ resid,
    float* __restrict__ outF, unsigned short* __restrict__ outB,
    int M, int N, int K)
{
    __shared__ unsigned short As[2][128][88];   // 176B pitch (16B aligned)
    __shared__ unsigned short Ws[2][64][144];   // 288B pitch (16B aligned)

    const int tid  = threadIdx.x;
    const int lane = tid & 31;
    const int wave = tid >> 5;
    const int hl   = lane >> 4;      // 16-lane group
    const int ln   = lane & 15;
    const int wm   = wave >> 2;      // 0..1  -> 64 rows each
    const int wn   = wave & 3;       // 0..3  -> 32 cols each
    const int bN0  = blockIdx.x * 128;
    const int bM0  = blockIdx.y * 128;

    v8f acc[4][2];
    const v8f vz = {0.f,0.f,0.f,0.f,0.f,0.f,0.f,0.f};
    #pragma unroll
    for (int mt = 0; mt < 4; ++mt)
        #pragma unroll
        for (int nt = 0; nt < 2; ++nt) acc[mt][nt] = vz;

    const int arow = tid >> 1, acol = (tid & 1) * 32;   // 128 rows x 64 cols
    const int wrow = tid >> 2, wcol = (tid & 3) * 32;   // 64 rows x 128 cols

    auto load_tiles = [&](int kb, int buf) {
        const unsigned short* gp = A + (size_t)(bM0 + arow) * K + kb + acol;
        #pragma unroll
        for (int j = 0; j < 4; ++j)
            async_cp16(&As[buf][arow][acol + j * 8], gp + j * 8);
        const unsigned short* wp = W + (size_t)(kb + wrow) * N + bN0 + wcol;
        #pragma unroll
        for (int j = 0; j < 4; ++j)
            async_cp16(&Ws[buf][wrow][wcol + j * 8], wp + j * 8);
    };

    load_tiles(0, 0);
    wait_async_lds();
    __syncthreads();

    int buf = 0;
    for (int kb = 0; kb < K; kb += 64) {
        if (kb + 64 < K) load_tiles(kb + 64, buf ^ 1);          // prefetch
        if (kb + 128 < K) {                                     // L2 prefetch
            __builtin_prefetch(A + (size_t)(bM0 + arow) * K + kb + 128 + acol, 0, 1);
            __builtin_prefetch(W + (size_t)(kb + 128 + wrow) * N + bN0 + wcol, 0, 1);
        }
        #pragma unroll
        for (int ks = 0; ks < 2; ++ks) {
            const int k0 = ks * 32;
            v16bf aF[4], bF[2];
            #pragma unroll
            for (int mt = 0; mt < 4; ++mt) {
                const int r = wm * 64 + mt * 16 + ln;
                #pragma unroll
                for (int i = 0; i < 16; ++i) {
                    int kk = k0 + hl * 16 + i;
                    aF[mt][i] = us2bf(As[buf][r][kk]);
                }
            }
            #pragma unroll
            for (int nt = 0; nt < 2; ++nt) {
                const int c = wn * 32 + nt * 16 + ln;
                #pragma unroll
                for (int i = 0; i < 16; ++i)
                    bF[nt][i] = us2bf(Ws[buf][k0 + hl * 16 + i][c]);
            }
            #pragma unroll
            for (int mt = 0; mt < 4; ++mt)
                #pragma unroll
                for (int nt = 0; nt < 2; ++nt)
                    acc[mt][nt] = __builtin_amdgcn_wmma_f32_16x16x32_bf16(
                        false, aF[mt], false, bF[nt], (short)0, acc[mt][nt],
                        false, false);
        }
        wait_async_lds();
        __syncthreads();
        buf ^= 1;
    }

    // epilogue
    #pragma unroll
    for (int mt = 0; mt < 4; ++mt) {
        #pragma unroll
        for (int nt = 0; nt < 2; ++nt) {
            const int gc = bN0 + wn * 32 + nt * 16 + ln;
            const float bs = bias[gc];
            #pragma unroll
            for (int v = 0; v < 8; ++v) {
                const int gr = bM0 + wm * 64 + mt * 16 + v + hl * 8;
                float val = acc[mt][nt][v] + bs;
                if (RELU) val = fmaxf(val, 0.0f);
                if (RES)  val += resid[(size_t)gr * N + gc];
                if (OUTBF) outB[(size_t)gr * N + gc] = f2bf(val);
                else       outF[(size_t)gr * N + gc] = val;
            }
        }
    }
}

// ---------- flash attention (bf16 in, bf16 ctx out, fp32 softmax) ----------
// grid: B*H*(S/128); block 256 = 8 waves; wave owns 16 query rows.
__global__ __launch_bounds__(256) void attn_wmma(
    const unsigned short* __restrict__ Q, const unsigned short* __restrict__ K,
    const unsigned short* __restrict__ V, unsigned short* __restrict__ O,
    int causal)
{
    __shared__ unsigned short Ks[32][88];
    __shared__ unsigned short Vs[32][88];
    __shared__ unsigned short Ps[8][16][40];

    const int tid  = threadIdx.x;
    const int lane = tid & 31;
    const int wave = tid >> 5;
    const int hl   = lane >> 4;
    const int ln   = lane & 15;
    const int nqb  = S_ / 128;
    const int qblk = blockIdx.x % nqb;
    const int h    = (blockIdx.x / nqb) % H_;
    const int b    = blockIdx.x / (nqb * H_);
    const int q0   = qblk * 128 + wave * 16;
    const size_t rb = (size_t)b * S_;
    const int hc   = h * DH_;

    // Q fragments (A layout), 2 chunks of K=32 over dh=64
    v16bf aQ[2];
    #pragma unroll
    for (int dh = 0; dh < 2; ++dh)
        #pragma unroll
        for (int i = 0; i < 16; ++i) {
            int kk = dh * 32 + hl * 16 + i;
            aQ[dh][i] = us2bf(Q[(rb + q0 + ln) * D_ + hc + kk]);
        }

    float mrun[8], lrun[8];
    #pragma unroll
    for (int v = 0; v < 8; ++v) { mrun[v] = -50.0f; lrun[v] = 0.0f; }
    const v8f vz = {0.f,0.f,0.f,0.f,0.f,0.f,0.f,0.f};
    v8f o[4];
    #pragma unroll
    for (int t = 0; t < 4; ++t) o[t] = vz;

    const int kend = causal ? (qblk * 128 + 128) : S_;
    const int lr = tid >> 3, lc = (tid & 7) * 8;   // 32x64 tile, 16B per thread

    for (int kv = 0; kv < kend; kv += 32) {
        __syncthreads();                             // prior consumers done
        async_cp16(&Ks[lr][lc], K + (rb + kv + lr) * D_ + hc + lc);
        async_cp16(&Vs[lr][lc], V + (rb + kv + lr) * D_ + hc + lc);
        wait_async_lds();
        __syncthreads();

        // scores: 16 q rows x 32 keys, two 16x16 C tiles
        float sc[2][8];
        #pragma unroll
        for (int nt = 0; nt < 2; ++nt) {
            v8f s = vz;
            #pragma unroll
            for (int dh = 0; dh < 2; ++dh) {
                v16bf bK;
                #pragma unroll
                for (int i = 0; i < 16; ++i)
                    bK[i] = us2bf(Ks[nt * 16 + ln][dh * 32 + hl * 16 + i]);
                s = __builtin_amdgcn_wmma_f32_16x16x32_bf16(
                        false, aQ[dh], false, bK, (short)0, s, false, false);
            }
            #pragma unroll
            for (int v = 0; v < 8; ++v) {
                float val = s[v] * 0.125f;           // 1/sqrt(64)
                int qg = q0 + v + hl * 8;
                int kg = kv + nt * 16 + ln;
                if (causal && (kg > qg)) val = -1e30f;
                sc[nt][v] = val;
            }
        }

        // online softmax per q row (reductions stay in 16-lane groups)
        float corr[8];
        #pragma unroll
        for (int v = 0; v < 8; ++v) {
            float m = fmaxf(sc[0][v], sc[1][v]);
            for (int off = 8; off >= 1; off >>= 1)
                m = fmaxf(m, __shfl_xor(m, off, 32));
            float mnew = fmaxf(mrun[v], m);
            float p0 = __expf(sc[0][v] - mnew);
            float p1 = __expf(sc[1][v] - mnew);
            float rs = p0 + p1;
            for (int off = 8; off >= 1; off >>= 1)
                rs += __shfl_xor(rs, off, 32);
            float cr = __expf(mrun[v] - mnew);
            lrun[v] = lrun[v] * cr + rs;
            mrun[v] = mnew;
            corr[v] = cr;
            const int rr = v + hl * 8;
            Ps[wave][rr][ln]      = f2bf(p0);
            Ps[wave][rr][16 + ln] = f2bf(p1);
        }
        #pragma unroll
        for (int t = 0; t < 4; ++t)
            #pragma unroll
            for (int v = 0; v < 8; ++v) o[t][v] *= corr[v];

        __syncthreads();   // make P visible cross-lane

        // P x V: A = P (16x32), B = V chunk (32x16) x4 over dh
        v16bf aP;
        #pragma unroll
        for (int i = 0; i < 16; ++i)
            aP[i] = us2bf(Ps[wave][ln][hl * 16 + i]);
        #pragma unroll
        for (int t = 0; t < 4; ++t) {
            v16bf bV;
            #pragma unroll
            for (int i = 0; i < 16; ++i)
                bV[i] = us2bf(Vs[hl * 16 + i][t * 16 + ln]);
            o[t] = __builtin_amdgcn_wmma_f32_16x16x32_bf16(
                       false, aP, false, bV, (short)0, o[t], false, false);
        }
    }

    #pragma unroll
    for (int t = 0; t < 4; ++t)
        #pragma unroll
        for (int v = 0; v < 8; ++v) {
            const size_t row = rb + q0 + v + hl * 8;
            O[row * D_ + hc + t * 16 + ln] = f2bf(o[t][v] / lrun[v]);
        }
}

// ------------------------------------------------------------------
extern "C" void kernel_launch(void* const* d_in, const int* in_sizes, int n_in,
                              void* d_out, int out_size, void* d_ws, size_t ws_size,
                              hipStream_t stream)
{
    const float* x     = (const float*)d_in[0];
    const float* srcx  = (const float*)d_in[1];
    const float* ln1g  = (const float*)d_in[4];
    const float* ln1b  = (const float*)d_in[5];
    const float* ln2g  = (const float*)d_in[6];
    const float* ln2b  = (const float*)d_in[7];
    const float* ln3g  = (const float*)d_in[8];
    const float* ln3b  = (const float*)d_in[9];
    const float* sa_wq = (const float*)d_in[10]; const float* sa_bq = (const float*)d_in[11];
    const float* sa_wk = (const float*)d_in[12]; const float* sa_bk = (const float*)d_in[13];
    const float* sa_wv = (const float*)d_in[14]; const float* sa_bv = (const float*)d_in[15];
    const float* sa_wo = (const float*)d_in[16]; const float* sa_bo = (const float*)d_in[17];
    const float* ca_wq = (const float*)d_in[18]; const float* ca_bq = (const float*)d_in[19];
    const float* ca_wk = (const float*)d_in[20]; const float* ca_bk = (const float*)d_in[21];
    const float* ca_wv = (const float*)d_in[22]; const float* ca_bv = (const float*)d_in[23];
    const float* ca_wo = (const float*)d_in[24]; const float* ca_bo = (const float*)d_in[25];
    const float* ff_w1 = (const float*)d_in[26]; const float* ff_b1 = (const float*)d_in[27];
    const float* ff_w2 = (const float*)d_in[28]; const float* ff_b2 = (const float*)d_in[29];
    float* out = (float*)d_out;

    // workspace layout (~144 MB)
    char* ws = (char*)d_ws;
    size_t off = 0;
    auto take = [&](size_t bytes) { void* p = ws + off; off += bytes; return p; };
    const size_t DD = (size_t)D_ * D_, DF = (size_t)D_ * F_, RD = (size_t)R_ * D_, RF = (size_t)R_ * F_;
    unsigned short* w_saq = (unsigned short*)take(DD * 2);
    unsigned short* w_sak = (unsigned short*)take(DD * 2);
    unsigned short* w_sav = (unsigned short*)take(DD * 2);
    unsigned short* w_sao = (unsigned short*)take(DD * 2);
    unsigned short* w_caq = (unsigned short*)take(DD * 2);
    unsigned short* w_cak = (unsigned short*)take(DD * 2);
    unsigned short* w_cav = (unsigned short*)take(DD * 2);
    unsigned short* w_cao = (unsigned short*)take(DD * 2);
    unsigned short* w_f1  = (unsigned short*)take(DF * 2);
    unsigned short* w_f2  = (unsigned short*)take(DF * 2);
    unsigned short* hb    = (unsigned short*)take(RD * 2);
    unsigned short* qb    = (unsigned short*)take(RD * 2);
    unsigned short* kb    = (unsigned short*)take(RD * 2);
    unsigned short* vb    = (unsigned short*)take(RD * 2);
    unsigned short* cb    = (unsigned short*)take(RD * 2);
    unsigned short* srcb  = (unsigned short*)take(RD * 2);
    float*          x1    = (float*)take(RD * 4);
    float*          x2    = (float*)take(RD * 4);
    unsigned short* ffb   = (unsigned short*)take(RF * 2);
    (void)ws_size; (void)in_sizes; (void)n_in; (void)out_size;

    auto conv = [&](const float* s, unsigned short* d, size_t n) {
        convert_f32_bf16<<<(unsigned)(n / 1024), 256, 0, stream>>>(s, d, (int)n);
    };
    conv(sa_wq, w_saq, DD); conv(sa_wk, w_sak, DD); conv(sa_wv, w_sav, DD); conv(sa_wo, w_sao, DD);
    conv(ca_wq, w_caq, DD); conv(ca_wk, w_cak, DD); conv(ca_wv, w_cav, DD); conv(ca_wo, w_cao, DD);
    conv(ff_w1, w_f1, DF);  conv(ff_w2, w_f2, DF);
    conv(srcx,  srcb, RD);

    const dim3 blk(256);
    auto ggrid = [](int M, int N) { return dim3(N / 128, M / 128); };
    const dim3 agrid(B_ * H_ * (S_ / 128));

    // ---- self attention ----
    layernorm_bf16<<<R_, blk, 0, stream>>>(x, ln1g, ln1b, hb);
    gemm_bf16_wmma<0,1,0><<<ggrid(R_, D_), blk, 0, stream>>>(hb, w_saq, sa_bq, nullptr, nullptr, qb, R_, D_, D_);
    gemm_bf16_wmma<0,1,0><<<ggrid(R_, D_), blk, 0, stream>>>(hb, w_sak, sa_bk, nullptr, nullptr, kb, R_, D_, D_);
    gemm_bf16_wmma<0,1,0><<<ggrid(R_, D_), blk, 0, stream>>>(hb, w_sav, sa_bv, nullptr, nullptr, vb, R_, D_, D_);
    attn_wmma<<<agrid, blk, 0, stream>>>(qb, kb, vb, cb, 1);
    gemm_bf16_wmma<0,0,1><<<ggrid(R_, D_), blk, 0, stream>>>(cb, w_sao, sa_bo, x, x1, nullptr, R_, D_, D_);

    // ---- cross attention (K/V from raw src_x) ----
    layernorm_bf16<<<R_, blk, 0, stream>>>(x1, ln2g, ln2b, hb);
    gemm_bf16_wmma<0,1,0><<<ggrid(R_, D_), blk, 0, stream>>>(hb,   w_caq, ca_bq, nullptr, nullptr, qb, R_, D_, D_);
    gemm_bf16_wmma<0,1,0><<<ggrid(R_, D_), blk, 0, stream>>>(srcb, w_cak, ca_bk, nullptr, nullptr, kb, R_, D_, D_);
    gemm_bf16_wmma<0,1,0><<<ggrid(R_, D_), blk, 0, stream>>>(srcb, w_cav, ca_bv, nullptr, nullptr, vb, R_, D_, D_);
    attn_wmma<<<agrid, blk, 0, stream>>>(qb, kb, vb, cb, 0);
    gemm_bf16_wmma<0,0,1><<<ggrid(R_, D_), blk, 0, stream>>>(cb, w_cao, ca_bo, x1, x2, nullptr, R_, D_, D_);

    // ---- FFN ----
    layernorm_bf16<<<R_, blk, 0, stream>>>(x2, ln3g, ln3b, hb);
    gemm_bf16_wmma<1,1,0><<<ggrid(R_, F_), blk, 0, stream>>>(hb,  w_f1, ff_b1, nullptr, nullptr, ffb, R_, F_, D_);
    gemm_bf16_wmma<0,0,1><<<ggrid(R_, D_), blk, 0, stream>>>(ffb, w_f2, ff_b2, x2, out, nullptr, R_, D_, F_);
}
